// GIN_basic_73512660239095
// MI455X (gfx1250) — matrix operand order repaired and checked by reading, not compiled, not run on hardware
//
#include <hip/hip_runtime.h>

#define N_NODES   20000
#define N_EDGES   256000
#define N_GRAPHS  512
#define DIM       300
#define HID       600
#define N_LAYERS  5
#define ATOM_VOCAB 128
#define BOND_VOCAB 16
#define NUM_CLASSES 10
#define BN_EPS 1e-5f
#define D4 (DIM / 4)

// padded GEMM geometry (all multiples of WMMA tile sizes)
#define MPAD  20096        // 157 * 128
#define KP1   320          // DIM  padded to mult of 32
#define NP1   608          // HID  padded to mult of 32
#define KP2   608          // HID  padded (K of GEMM2)
#define NP2   320          // DIM  padded (N of GEMM2)

#define USE_TDM 1          // stage GEMM tiles with tensor_load_to_lds (TDM)

typedef __attribute__((ext_vector_type(16))) __bf16 v16bf;
typedef __attribute__((ext_vector_type(8)))  float v8f;
typedef __attribute__((ext_vector_type(4)))  unsigned int u32x4;
typedef __attribute__((ext_vector_type(8)))  int i32x8;
typedef __attribute__((ext_vector_type(4)))  int i32x4;

__device__ __forceinline__ unsigned short f2bf(float f) {
  unsigned int u = __float_as_uint(f);
  u += 0x7FFFu + ((u >> 16) & 1u);   // round-to-nearest-even truncation
  return (unsigned short)(u >> 16);
}

__device__ __forceinline__ void atomic_fadd(float* p, float v) {
  __hip_atomic_fetch_add(p, v, __ATOMIC_RELAXED, __HIP_MEMORY_SCOPE_AGENT);
}

#if USE_TDM
// One 2D TDM descriptor: move rows x row_elems (bf16, 2B) from global
// (row stride = ld elems) into LDS at lds_addr (packed tile). ISA 8.3/8.4.
__device__ __forceinline__ void tdm_load_2d_bf16(
    unsigned lds_addr, const void* gaddr,
    unsigned rows, unsigned row_elems, unsigned ld) {
  unsigned long long ga = (unsigned long long)(uintptr_t)gaddr;
  u32x4 g0;
  g0[0] = 1u;                                       // count=1, user mode
  g0[1] = lds_addr;                                 // D#.lds_addr
  g0[2] = (unsigned)(ga & 0xFFFFFFFFu);             // global_addr[31:0]
  g0[3] = (unsigned)((ga >> 32) & 0x01FFFFFFu)      // global_addr[56:32]
        | (2u << 30);                               // type = 2 ("image")
  unsigned tw = ld;                                 // tensor_dim0 (x bound)
  unsigned th = 0x00100000u;                        // tensor_dim1 (large)
  i32x8 g1;
  g1[0] = (int)(1u << 16);                          // data_size = 1 -> 2 bytes
  g1[1] = (int)((tw & 0xFFFFu) << 16);              // tensor_dim0[15:0]
  g1[2] = (int)((tw >> 16) | ((th & 0xFFFFu) << 16));
  g1[3] = (int)((th >> 16) | (row_elems << 16));    // tile_dim0
  g1[4] = (int)rows;                                // tile_dim1 (tile_dim2=0)
  g1[5] = (int)ld;                                  // tensor_dim0_stride lo
  g1[6] = 0;
  g1[7] = 0;
  i32x4 gz = {0, 0, 0, 0};
#if defined(__clang_major__) && (__clang_major__ >= 23)
  i32x8 gz8 = {0, 0, 0, 0, 0, 0, 0, 0};
  __builtin_amdgcn_tensor_load_to_lds(g0, g1, gz, gz, gz8, 0);
#else
  __builtin_amdgcn_tensor_load_to_lds(g0, g1, gz, gz, 0);
#endif
}
#endif

// ---------------------------------------------------------------- encoders
__global__ __launch_bounds__(256) void atom_encoder_kernel(
    const int* __restrict__ x, const float* __restrict__ emb,
    float* __restrict__ h) {
  int idx = blockIdx.x * blockDim.x + threadIdx.x;   // over N * D4
  if (idx >= N_NODES * D4) return;
  int n = idx / D4, d4 = idx - n * D4;
  float4 acc = make_float4(0.f, 0.f, 0.f, 0.f);
#pragma unroll
  for (int f = 0; f < 9; ++f) {
    int xi = x[n * 9 + f];
    const float4* p = (const float4*)(emb + (size_t)(f * ATOM_VOCAB + xi) * DIM);
    float4 v = p[d4];
    acc.x += v.x; acc.y += v.y; acc.z += v.z; acc.w += v.w;
  }
  ((float4*)(h + (size_t)n * DIM))[d4] = acc;
}

__global__ __launch_bounds__(256) void bond_encoder_kernel(
    const int* __restrict__ ea, const float* __restrict__ emb,
    float* __restrict__ e) {
  int idx = blockIdx.x * blockDim.x + threadIdx.x;   // over E * D4
  if (idx >= N_EDGES * D4) return;
  int ei = idx / D4, d4 = idx - ei * D4;
  float4 acc = make_float4(0.f, 0.f, 0.f, 0.f);
#pragma unroll
  for (int f = 0; f < 3; ++f) {
    int bi = ea[ei * 3 + f];
    const float4* p = (const float4*)(emb + (size_t)(f * BOND_VOCAB + bi) * DIM);
    float4 v = p[d4];
    acc.x += v.x; acc.y += v.y; acc.z += v.z; acc.w += v.w;
  }
  ((float4*)(e + (size_t)ei * DIM))[d4] = acc;
}

// ---------------------------------------------------------------- utility
__global__ __launch_bounds__(256) void copy4_kernel(
    const float* __restrict__ in, float* __restrict__ out, int n4) {
  int i = blockIdx.x * blockDim.x + threadIdx.x;
  if (i < n4) ((float4*)out)[i] = ((const float4*)in)[i];
}

__global__ __launch_bounds__(256) void zero_kernel(float* __restrict__ p, int n) {
  int i = blockIdx.x * blockDim.x + threadIdx.x;
  if (i < n) p[i] = 0.f;
}

__global__ __launch_bounds__(256) void zero_u4_kernel(uint4* __restrict__ p, int n) {
  int i = blockIdx.x * blockDim.x + threadIdx.x;
  if (i < n) p[i] = make_uint4(0u, 0u, 0u, 0u);
}

// ------------------------------------------------ bf16 conversion (padded)
// z[N,300] f32 -> zb[MPAD,320] bf16 (pad K cols zeroed; rows>=N untouched)
__global__ __launch_bounds__(256) void convA_kernel(
    const float* __restrict__ src, unsigned int* __restrict__ dst) {
  const int CW = KP1 / 2;                       // 160 packed uints per row
  int idx = blockIdx.x * blockDim.x + threadIdx.x;
  if (idx >= N_NODES * CW) return;
  int n = idx / CW, c2 = idx - n * CW;
  int c = 2 * c2;
  float v0 = (c     < DIM) ? src[(size_t)n * DIM + c]     : 0.f;
  float v1 = (c + 1 < DIM) ? src[(size_t)n * DIM + c + 1] : 0.f;
  dst[(size_t)n * CW + c2] = (unsigned)f2bf(v0) | ((unsigned)f2bf(v1) << 16);
}

// W1[L,300,600] f32 -> W1t[L,608,320] bf16, TRANSPOSED ([n][k]) zero-padded
__global__ __launch_bounds__(256) void convW1t_kernel(
    const float* __restrict__ src, unsigned int* __restrict__ dst) {
  const int CW = KP1 / 2;                       // 160 uints per n-row
  int idx = blockIdx.x * blockDim.x + threadIdx.x;
  if (idx >= N_LAYERS * NP1 * CW) return;
  int l = idx / (NP1 * CW), rem = idx - l * (NP1 * CW);
  int n = rem / CW, k2 = rem - n * CW;
  int c = 2 * k2;
  float v0 = (n < HID && c     < DIM) ? src[((size_t)l * DIM + c)     * HID + n] : 0.f;
  float v1 = (n < HID && c + 1 < DIM) ? src[((size_t)l * DIM + c + 1) * HID + n] : 0.f;
  dst[idx] = (unsigned)f2bf(v0) | ((unsigned)f2bf(v1) << 16);
}

// W2[L,600,300] f32 -> W2t[L,320,608] bf16, TRANSPOSED ([n][k]) zero-padded
__global__ __launch_bounds__(256) void convW2t_kernel(
    const float* __restrict__ src, unsigned int* __restrict__ dst) {
  const int CW = KP2 / 2;                       // 304 uints per n-row
  int idx = blockIdx.x * blockDim.x + threadIdx.x;
  if (idx >= N_LAYERS * NP2 * CW) return;
  int l = idx / (NP2 * CW), rem = idx - l * (NP2 * CW);
  int n = rem / CW, k2 = rem - n * CW;
  int c = 2 * k2;
  float v0 = (n < DIM && c     < HID) ? src[((size_t)l * HID + c)     * DIM + n] : 0.f;
  float v1 = (n < DIM && c + 1 < HID) ? src[((size_t)l * HID + c + 1) * DIM + n] : 0.f;
  dst[idx] = (unsigned)f2bf(v0) | ((unsigned)f2bf(v1) << 16);
}

// --------------------------------------------------- edge message + scatter
// z[dst] += relu(h[src] + e)   (z pre-initialized with h)
__global__ __launch_bounds__(256) void edge_scatter_kernel(
    const float* __restrict__ h, const float* __restrict__ e,
    const int* __restrict__ src, const int* __restrict__ dst,
    float* __restrict__ z) {
  int idx = blockIdx.x * blockDim.x + threadIdx.x;   // over E * D4
  if (idx >= N_EDGES * D4) return;
  int ei = idx / D4, d4 = idx - ei * D4;
  int s = src[ei], t = dst[ei];
  float4 hv = ((const float4*)(h + (size_t)s * DIM))[d4];
  float4 ev = ((const float4*)(e + (size_t)ei * DIM))[d4];
  float* zp = z + (size_t)t * DIM + 4 * d4;
  atomic_fadd(zp + 0, fmaxf(hv.x + ev.x, 0.f));
  atomic_fadd(zp + 1, fmaxf(hv.y + ev.y, 0.f));
  atomic_fadd(zp + 2, fmaxf(hv.z + ev.z, 0.f));
  atomic_fadd(zp + 3, fmaxf(hv.w + ev.w, 0.f));
}

// --------------------------------------------------- WMMA GEMM + BN (+ReLU)
// out = bn(A_bf16[MPAD,Kp] @ Wt_bf16[Ncp,Kp]^T + bias) ; out f32 OR bf16
// Wt is pre-transposed: row n holds K-contiguous weights.
// Block: 256 threads = 8 waves; block tile 128(M) x 32(N); K staged 32-wide.
__global__ __launch_bounds__(256) void gemm_bf16_kernel(
    const unsigned short* __restrict__ A, int lda,
    const unsigned short* __restrict__ Wt, int ldw,
    const float* __restrict__ bias,
    const float* __restrict__ gamma, const float* __restrict__ beta,
    const float* __restrict__ mean,  const float* __restrict__ var,
    float* __restrict__ outf, unsigned short* __restrict__ outh, int ldo,
    int Kp, int NcReal, int relu_flag) {
  __shared__ __align__(16) unsigned short As[128][32];  // 8 KB
  __shared__ __align__(16) unsigned short Bs[32][32];   // 2 KB  [n][k]

  const int tid  = threadIdx.x;
  const int wave = tid >> 5;
  const int lane = tid & 31;
  const int hi   = lane >> 4;
  const int l16  = lane & 15;
  const int tileM = blockIdx.x * 128;
  const int tileN = blockIdx.y * 32;

  v8f acc0 = {}, acc1 = {};

#if USE_TDM
  const unsigned asLds = (unsigned)(uintptr_t)(void*)&As[0][0];
  const unsigned bsLds = (unsigned)(uintptr_t)(void*)&Bs[0][0];
#else
  const int arow  = tid >> 1;          // 2 threads per A row
  const int ahalf = tid & 1;           // each stages 32 bytes (2 x uint4)
  const unsigned short* Arow = A + (size_t)(tileM + arow) * lda;
#endif

  for (int k0 = 0; k0 < Kp; k0 += 32) {
#if USE_TDM
    if (wave == 0) {
      tdm_load_2d_bf16(asLds, A  + (size_t)tileM * lda + k0, 128, 32, lda);
      tdm_load_2d_bf16(bsLds, Wt + (size_t)tileN * ldw + k0,  32, 32, ldw);
      __builtin_amdgcn_s_wait_tensorcnt(0);
    }
    __syncthreads();
#else
    {   // ---- stage A tile (128x32 bf16): pure b128 traffic
      const uint4* srcp = (const uint4*)(Arow + k0);
      uint4 a0 = srcp[ahalf * 2 + 0];
      uint4 a1 = srcp[ahalf * 2 + 1];
      uint4* dstp = (uint4*)&As[arow][0];
      dstp[ahalf * 2 + 0] = a0;
      dstp[ahalf * 2 + 1] = a1;
      if (k0 + 32 < Kp) __builtin_prefetch(Arow + k0 + 32, 0, 1);
    }
    if (tid < 128) {   // ---- stage B tile from pre-transposed weights
      int n = tid >> 2, seg = tid & 3;
      const uint4* sp = (const uint4*)(Wt + (size_t)(tileN + n) * ldw + k0);
      uint4 bv = sp[seg];
      ((uint4*)&Bs[n][0])[seg] = bv;
    }
    __syncthreads();
#endif

    // ---- fragments per gfx1250 wave32 16-bit layouts
    union { v16bf v; unsigned int u[8]; } afrag, bfrag0, bfrag1;
#pragma unroll
    for (int vv = 0; vv < 8; ++vv) {
      int kA = (vv < 4 ? 2 * vv : 16 + 2 * (vv - 4)) + hi * 8;
      afrag.u[vv]  = *(const unsigned int*)&As[wave * 16 + l16][kA];
      int kB = (vv < 4 ? 2 * vv : 16 + 2 * (vv - 4)) + hi * 8;
      bfrag0.u[vv] = *(const unsigned int*)&Bs[l16][kB];
      bfrag1.u[vv] = *(const unsigned int*)&Bs[16 + l16][kB];
    }
    acc0 = __builtin_amdgcn_wmma_f32_16x16x32_bf16(
        false, afrag.v, false, bfrag0.v, (short)0, acc0, false, false);
    acc1 = __builtin_amdgcn_wmma_f32_16x16x32_bf16(
        false, afrag.v, false, bfrag1.v, (short)0, acc1, false, false);
    __syncthreads();
  }

  // ---- epilogue: D VGPR r -> (m = r + 8*hi, n = lane&15)
#pragma unroll
  for (int j = 0; j < 2; ++j) {
    int n = tileN + 16 * j + l16;
    if (n < NcReal) {
      float scale = gamma[n] * rsqrtf(var[n] + BN_EPS);
      float mu = mean[n], bt = beta[n], bs = bias[n];
#pragma unroll
      for (int r = 0; r < 8; ++r) {
        int m = tileM + wave * 16 + r + 8 * hi;
        if (m < N_NODES) {
          float a = j ? acc1[r] : acc0[r];
          float y = (a + bs - mu) * scale + bt;
          if (relu_flag) y = fmaxf(y, 0.f);
          if (outh) outh[(size_t)m * ldo + n] = f2bf(y);
          else      outf[(size_t)m * ldo + n] = y;
        }
      }
    }
  }
}

// ---------------------------------------------------------------- pooling
__global__ __launch_bounds__(256) void pool_kernel(
    const float* __restrict__ h, const int* __restrict__ batch,
    float* __restrict__ pool) {
  int idx = blockIdx.x * blockDim.x + threadIdx.x;   // over N * D4
  if (idx >= N_NODES * D4) return;
  int n = idx / D4, d4 = idx - n * D4;
  int g = batch[n];
  float4 v = ((const float4*)(h + (size_t)n * DIM))[d4];
  float* pp = pool + (size_t)g * DIM + 4 * d4;
  atomic_fadd(pp + 0, v.x);
  atomic_fadd(pp + 1, v.y);
  atomic_fadd(pp + 2, v.z);
  atomic_fadd(pp + 3, v.w);
}

__global__ __launch_bounds__(256) void count_kernel(
    const int* __restrict__ batch, float* __restrict__ cnt) {
  int n = blockIdx.x * blockDim.x + threadIdx.x;
  if (n < N_NODES) atomic_fadd(&cnt[batch[n]], 1.f);
}

__global__ __launch_bounds__(256) void pred_kernel(
    const float* __restrict__ pool, const float* __restrict__ cnt,
    const float* __restrict__ Wp, const float* __restrict__ bp,
    float* __restrict__ out) {
  int idx = blockIdx.x * blockDim.x + threadIdx.x;   // over G * NUM_CLASSES
  if (idx >= N_GRAPHS * NUM_CLASSES) return;
  int g = idx / NUM_CLASSES, c = idx - g * NUM_CLASSES;
  float inv = 1.f / fmaxf(cnt[g], 1.f);
  float acc = bp[c];
  for (int d = 0; d < DIM; ++d)
    acc += pool[(size_t)g * DIM + d] * inv * Wp[d * NUM_CLASSES + c];
  out[idx] = acc;
}

// ---------------------------------------------------------------- launcher
extern "C" void kernel_launch(void* const* d_in, const int* in_sizes, int n_in,
                              void* d_out, int out_size, void* d_ws, size_t ws_size,
                              hipStream_t stream) {
  const int*   x         = (const int*)d_in[0];
  const int*   edge_idx  = (const int*)d_in[1];
  const int*   edge_attr = (const int*)d_in[2];
  const int*   batch     = (const int*)d_in[3];
  const float* atom_emb  = (const float*)d_in[4];
  const float* bond_emb  = (const float*)d_in[5];
  const float* W1        = (const float*)d_in[6];
  const float* b1        = (const float*)d_in[7];
  const float* bn1_g     = (const float*)d_in[8];
  const float* bn1_b     = (const float*)d_in[9];
  const float* bn1_m     = (const float*)d_in[10];
  const float* bn1_v     = (const float*)d_in[11];
  const float* W2        = (const float*)d_in[12];
  const float* b2        = (const float*)d_in[13];
  const float* bn2_g     = (const float*)d_in[14];
  const float* bn2_b     = (const float*)d_in[15];
  const float* bn2_m     = (const float*)d_in[16];
  const float* bn2_v     = (const float*)d_in[17];
  const float* pred_W    = (const float*)d_in[18];
  const float* pred_b    = (const float*)d_in[19];
  float* out = (float*)d_out;

  // workspace carve-out
  char* base = (char*)d_ws;
  size_t off = 0;
  auto carve = [&](size_t bytes) {
    void* p = base + off;
    off += (bytes + 255) & ~(size_t)255;
    return p;
  };
  float*          h     = (float*)carve((size_t)N_NODES * DIM * 4);
  float*          e     = (float*)carve((size_t)N_EDGES * DIM * 4);
  float*          z     = (float*)carve((size_t)N_NODES * DIM * 4);
  unsigned short* zb    = (unsigned short*)carve((size_t)MPAD * KP1 * 2);
  unsigned short* buf1b = (unsigned short*)carve((size_t)MPAD * NP1 * 2);
  unsigned short* w1t   = (unsigned short*)carve((size_t)N_LAYERS * NP1 * KP1 * 2);
  unsigned short* w2t   = (unsigned short*)carve((size_t)N_LAYERS * NP2 * KP2 * 2);
  float*          pool  = (float*)carve((size_t)N_GRAPHS * DIM * 4 + N_GRAPHS * 4);
  float*          cnt   = pool + (size_t)N_GRAPHS * DIM;

  const int* src = edge_idx;
  const int* dst = edge_idx + N_EDGES;

  const int TPB = 256;
  int nd4 = N_NODES * D4;
  int ed4 = N_EDGES * D4;

  // one-time (per call) weight transpose/convert + pad-zeroing of buf1b
  {
    int w1u = N_LAYERS * NP1 * (KP1 / 2);
    int w2u = N_LAYERS * NP2 * (KP2 / 2);
    convW1t_kernel<<<(w1u + TPB - 1) / TPB, TPB, 0, stream>>>(W1, (unsigned int*)w1t);
    convW2t_kernel<<<(w2u + TPB - 1) / TPB, TPB, 0, stream>>>(W2, (unsigned int*)w2t);
    int b1q = (int)(((size_t)MPAD * NP1 * 2) / 16);
    zero_u4_kernel<<<(b1q + TPB - 1) / TPB, TPB, 0, stream>>>((uint4*)buf1b, b1q);
  }

  atom_encoder_kernel<<<(nd4 + TPB - 1) / TPB, TPB, 0, stream>>>(x, atom_emb, h);
  bond_encoder_kernel<<<(ed4 + TPB - 1) / TPB, TPB, 0, stream>>>(edge_attr, bond_emb, e);

  for (int l = 0; l < N_LAYERS; ++l) {
    // z = h  (self term, eps=0), then z[dst] += relu(h[src] + e)
    copy4_kernel<<<(nd4 + TPB - 1) / TPB, TPB, 0, stream>>>(h, z, nd4);
    edge_scatter_kernel<<<(ed4 + TPB - 1) / TPB, TPB, 0, stream>>>(h, e, src, dst, z);
    // zb = bf16(z) padded
    int zu = N_NODES * (KP1 / 2);
    convA_kernel<<<(zu + TPB - 1) / TPB, TPB, 0, stream>>>(z, (unsigned int*)zb);
    // buf1b = bf16(relu(bn1(zb @ W1 + b1)))
    {
      dim3 grid(MPAD / 128, NP1 / 32);
      gemm_bf16_kernel<<<grid, TPB, 0, stream>>>(
          zb, KP1, w1t + (size_t)l * NP1 * KP1, KP1,
          b1 + (size_t)l * HID,
          bn1_g + (size_t)l * HID, bn1_b + (size_t)l * HID,
          bn1_m + (size_t)l * HID, bn1_v + (size_t)l * HID,
          nullptr, buf1b, NP1, KP1, HID, 1);
    }
    // h = bn2(buf1b @ W2 + b2)  [+relu except last layer]
    {
      dim3 grid(MPAD / 128, NP2 / 32);
      gemm_bf16_kernel<<<grid, TPB, 0, stream>>>(
          buf1b, NP1, w2t + (size_t)l * NP2 * KP2, KP2,
          b2 + (size_t)l * DIM,
          bn2_g + (size_t)l * DIM, bn2_b + (size_t)l * DIM,
          bn2_m + (size_t)l * DIM, bn2_v + (size_t)l * DIM,
          h, nullptr, DIM, KP2, DIM, (l < N_LAYERS - 1) ? 1 : 0);
    }
  }

  // global mean pool + prediction head
  int pz = N_GRAPHS * DIM + N_GRAPHS;
  zero_kernel<<<(pz + TPB - 1) / TPB, TPB, 0, stream>>>(pool, pz);
  pool_kernel<<<(nd4 + TPB - 1) / TPB, TPB, 0, stream>>>(h, batch, pool);
  count_kernel<<<(N_NODES + TPB - 1) / TPB, TPB, 0, stream>>>(batch, cnt);
  pred_kernel<<<(N_GRAPHS * NUM_CLASSES + TPB - 1) / TPB, TPB, 0, stream>>>(
      pool, cnt, pred_W, pred_b, out);
}
